// GatedAttentionMIL_54546084659650
// MI455X (gfx1250) — compile-verified, matrix operand order checked by reading, not run on hardware
//
#include <hip/hip_runtime.h>
#include <hip/hip_bf16.h>
#include <math.h>

// ---------------------------------------------------------------------------
// GatedAttentionMIL on MI455X (gfx1250, wave32, WMMA)
//   features [16,4096,1024] f32, Wv/Wu [1024,256], bv/bu [256], Ww [256,1], bw [1]
//   out = concat(aggregated [16,1024], weights [16,4096])
// ---------------------------------------------------------------------------

typedef __attribute__((ext_vector_type(16))) __bf16 v16bf;
typedef __attribute__((ext_vector_type(8)))  float  v8f;

#define D_DIM 1024
#define H_DIM 256
#define KROWS 4096
#define BATCH 16
#define ROWS_TOTAL (BATCH * KROWS)          // 65536
#define W_ELEMS (D_DIM * H_DIM)             // 262144 per matrix
#define ROWS_PER_BLOCK 64                   // 4 row-tiles of 16
#define A_LDS_ELEMS (ROWS_PER_BLOCK * D_DIM)            // 65536 bf16 = 128 KB
#define GEMM_SMEM_BYTES (A_LDS_ELEMS * 2 + ROWS_PER_BLOCK * 4)

#if __has_builtin(__builtin_amdgcn_sched_group_barrier)
#define SGB(mask, size, id) __builtin_amdgcn_sched_group_barrier((mask), (size), (id))
#else
#define SGB(mask, size, id)
#endif

// fp32 -> bf16 with round-to-nearest-even
__device__ __forceinline__ unsigned short f32_to_bf16(float f) {
    unsigned u = __float_as_uint(f);
    unsigned r = u + 0x7FFFu + ((u >> 16) & 1u);
    return (unsigned short)(r >> 16);
}

// ISA 16-bit fragment striping (A 16x32 / B 32x16 mirrored):
//   kk in [0,8)   -> lane = idx,    i = kk
//   kk in [8,16)  -> lane = idx+16, i = kk-8
//   kk in [16,24) -> lane = idx,    i = kk-8
//   kk in [24,32) -> lane = idx+16, i = kk-16
__device__ __forceinline__ void frag_slot(int idx, int kk, int& lane, int& i) {
    if (kk < 8)       { lane = idx;      i = kk;      }
    else if (kk < 16) { lane = idx + 16; i = kk - 8;  }
    else if (kk < 24) { lane = idx;      i = kk - 8;  }
    else              { lane = idx + 16; i = kk - 16; }
}

// ---------------------------------------------------------------------------
// Kernel 0: swizzle Wv/Wu (f32 row-major [D][H]) into bf16 B-fragment layout
// in workspace; also zero the aggregated-output region of d_out.
//   frag(ht, ds): 32 lanes x 16 bf16; off = ((ht*32 + ds)*32 + lane)*16 + i
// ---------------------------------------------------------------------------
__global__ void prep_kernel(const float* __restrict__ Wv,
                            const float* __restrict__ Wu,
                            unsigned short* __restrict__ wvS,
                            unsigned short* __restrict__ wuS,
                            float* __restrict__ agg_out) {
    int tid = blockIdx.x * blockDim.x + threadIdx.x;   // 0 .. 2*W_ELEMS-1
    if (tid < BATCH * D_DIM) agg_out[tid] = 0.0f;      // zero aggregated region
    const float* src = (tid < W_ELEMS) ? Wv : Wu;
    unsigned short* dst = (tid < W_ELEMS) ? wvS : wuS;
    int e = tid & (W_ELEMS - 1);                       // d*H + h
    int d = e >> 8;
    int h = e & 255;
    int ht = h >> 4, n = h & 15;
    int ds = d >> 5, kk = d & 31;
    int lane, i;
    frag_slot(n, kk, lane, i);
    dst[(((ht * 32 + ds) * 32 + lane) << 4) + i] = f32_to_bf16(src[e]);
}

// ---------------------------------------------------------------------------
// Kernel 1: WMMA GEMM + gate + H->1 projection -> logits[row]
// Block = 256 threads (8 waves), owns 64 rows (4 row-tiles).
// Software-pipelined inner loop: while the 8-WMMA chain for iteration i
// executes, the 8 ds_load_b128 + 4 global_load_b128 for iteration i+1 are
// issued into fresh registers. sched_group_barrier pins the interleave
// (WMMA x2, DS-read x2) x4, VMEM-read x4 so the scheduler cannot collapse
// the fragment sets back into one register set (which caused dscnt-0 waits
// and WMMA WAR-hazard v_nops in previous rounds).
// ---------------------------------------------------------------------------
__global__ void __launch_bounds__(256, 2)
gemm_logits_kernel(const float* __restrict__ feat,
                   const unsigned short* __restrict__ wvS,
                   const unsigned short* __restrict__ wuS,
                   const float* __restrict__ bv,
                   const float* __restrict__ bu,
                   const float* __restrict__ Ww,
                   const float* __restrict__ bw,
                   float* __restrict__ logits) {
    extern __shared__ __attribute__((aligned(32))) unsigned char smem[];
    unsigned short* ldsA = (unsigned short*)smem;                 // 128 KB
    float* ldsLogit = (float*)(smem + A_LDS_ELEMS * 2);           // 64 floats

    const int tid  = threadIdx.x;
    const int lane = tid & 31;
    const int wave = tid >> 5;
    const int rowBase = blockIdx.x * ROWS_PER_BLOCK;

    if (tid < ROWS_PER_BLOCK) ldsLogit[tid] = 0.0f;

    // ---- stage A: float4 loads, pack 4x bf16, swizzled 8B LDS stores -------
    for (int e4 = tid; e4 < (ROWS_PER_BLOCK * D_DIM) / 4; e4 += 256) {
        int e = e4 * 4;
        int r = e >> 10;                 // block-local row 0..63
        int c = e & (D_DIM - 1);         // col, multiple of 4
        const float4 f = *reinterpret_cast<const float4*>(
            feat + (size_t)(rowBase + r) * D_DIM + c);
        unsigned p0 = (unsigned)f32_to_bf16(f.x) | ((unsigned)f32_to_bf16(f.y) << 16);
        unsigned p1 = (unsigned)f32_to_bf16(f.z) | ((unsigned)f32_to_bf16(f.w) << 16);
        int rt = r >> 4, m = r & 15;
        int ds = c >> 5, kk = c & 31;
        int l, i;
        frag_slot(m, kk, l, i);
        uint2* dst = reinterpret_cast<uint2*>(
            &ldsA[rt * (16 * D_DIM) + ds * 512 + (l << 4) + i]);
        *dst = make_uint2(p0, p1);
    }
    __syncthreads();

    const unsigned short* aBase = ldsA + (lane << 4);

#define LA(rt, dsx) (*reinterpret_cast<const v16bf*>(aBase + (rt) * (16 * D_DIM) + (dsx) * 512))

    // ---- per-wave: two h-tiles, four row-tiles -----------------------------
    for (int ht = wave * 2; ht < wave * 2 + 2; ++ht) {
        const int n = ht * 16 + (lane & 15);           // output column in H
        const float bvn = bv[n];
        const float bun = bu[n];
        const float wwn = Ww[n];

        v8f accV0 = {}, accV1 = {}, accV2 = {}, accV3 = {};
        v8f accU0 = {}, accU1 = {}, accU2 = {}, accU3 = {};
        const unsigned short* bvBase = wvS + (((size_t)ht * 32) << 9) + (lane << 4);
        const unsigned short* buBase = wuS + (((size_t)ht * 32) << 9) + (lane << 4);

        // prologue: fragment set for ds = 0
        v16bf a0 = LA(0, 0), a1 = LA(1, 0), a2 = LA(2, 0), a3 = LA(3, 0);
        v16bf bV = *reinterpret_cast<const v16bf*>(bvBase);
        v16bf bU = *reinterpret_cast<const v16bf*>(buBase);

        #pragma unroll 2
        for (int ds = 0; ds < 32; ++ds) {
            const int nx = (ds + 1) & 31;   // wrap: last preload is dead but harmless
            // next-iteration loads into fresh values
            v16bf na0 = LA(0, nx), na1 = LA(1, nx), na2 = LA(2, nx), na3 = LA(3, nx);
            v16bf nbV = *reinterpret_cast<const v16bf*>(bvBase + (nx << 9));
            v16bf nbU = *reinterpret_cast<const v16bf*>(buBase + (nx << 9));
            // current-iteration 8-WMMA chain
            accV0 = __builtin_amdgcn_wmma_f32_16x16x32_bf16(false, a0, false, bV, (short)0, accV0, false, false);
            accU0 = __builtin_amdgcn_wmma_f32_16x16x32_bf16(false, a0, false, bU, (short)0, accU0, false, false);
            accV1 = __builtin_amdgcn_wmma_f32_16x16x32_bf16(false, a1, false, bV, (short)0, accV1, false, false);
            accU1 = __builtin_amdgcn_wmma_f32_16x16x32_bf16(false, a1, false, bU, (short)0, accU1, false, false);
            accV2 = __builtin_amdgcn_wmma_f32_16x16x32_bf16(false, a2, false, bV, (short)0, accV2, false, false);
            accU2 = __builtin_amdgcn_wmma_f32_16x16x32_bf16(false, a2, false, bU, (short)0, accU2, false, false);
            accV3 = __builtin_amdgcn_wmma_f32_16x16x32_bf16(false, a3, false, bV, (short)0, accV3, false, false);
            accU3 = __builtin_amdgcn_wmma_f32_16x16x32_bf16(false, a3, false, bU, (short)0, accU3, false, false);
            // pin steady-state interleave: (WMMA x2, DS-read x2) x4, VMEM-read x4
            SGB(0x008, 2, 0); SGB(0x100, 2, 0);
            SGB(0x008, 2, 0); SGB(0x100, 2, 0);
            SGB(0x008, 2, 0); SGB(0x100, 2, 0);
            SGB(0x008, 2, 0); SGB(0x100, 2, 0);
            SGB(0x020, 4, 0);
            // rotate
            a0 = na0; a1 = na1; a2 = na2; a3 = na3; bV = nbV; bU = nbU;
        }

        // gate, project onto Ww, reduce across the 16 lanes holding one row
        v8f accVs[4] = { accV0, accV1, accV2, accV3 };
        v8f accUs[4] = { accU0, accU1, accU2, accU3 };
        #pragma unroll
        for (int rt = 0; rt < 4; ++rt) {
            #pragma unroll
            for (int j = 0; j < 8; ++j) {
                float v = tanhf(accVs[rt][j] + bvn);
                float u = 1.0f / (1.0f + expf(-(accUs[rt][j] + bun)));
                float g = v * u * wwn;
                g += __shfl_xor(g, 8, 32);
                g += __shfl_xor(g, 4, 32);
                g += __shfl_xor(g, 2, 32);
                g += __shfl_xor(g, 1, 32);
                if ((lane & 15) == 0) {
                    int row = rt * 16 + j + ((lane >> 4) << 3);
                    atomicAdd(&ldsLogit[row], g);      // ds_add_f32
                }
            }
        }
    }
#undef LA
    __syncthreads();

    if (tid < ROWS_PER_BLOCK) logits[rowBase + tid] = ldsLogit[tid] + bw[0];
}

// ---------------------------------------------------------------------------
// Kernel 2: per-batch softmax over K=4096 logits -> weights (d_out region)
// ---------------------------------------------------------------------------
__global__ void softmax_kernel(const float* __restrict__ logits,
                               float* __restrict__ weights) {
    __shared__ float red[256];
    const int b = blockIdx.x, tid = threadIdx.x;
    const float* lb = logits + (size_t)b * KROWS;

    float lv[16];
    float m = -INFINITY;
    #pragma unroll
    for (int j = 0; j < 16; ++j) {
        lv[j] = lb[tid + j * 256];
        m = fmaxf(m, lv[j]);
    }
    red[tid] = m;
    __syncthreads();
    for (int s = 128; s > 0; s >>= 1) {
        if (tid < s) red[tid] = fmaxf(red[tid], red[tid + s]);
        __syncthreads();
    }
    const float M = red[0];
    __syncthreads();

    float sum = 0.0f;
    #pragma unroll
    for (int j = 0; j < 16; ++j) {
        lv[j] = expf(lv[j] - M);
        sum += lv[j];
    }
    red[tid] = sum;
    __syncthreads();
    for (int s = 128; s > 0; s >>= 1) {
        if (tid < s) red[tid] += red[tid + s];
        __syncthreads();
    }
    const float inv = 1.0f / red[0];

    float* wb = weights + (size_t)b * KROWS;
    #pragma unroll
    for (int j = 0; j < 16; ++j) wb[tid + j * 256] = lv[j] * inv;
}

// ---------------------------------------------------------------------------
// Kernel 3: aggregated[b,d] = sum_k feat[b,k,d] * w[b,k]
// grid (B, 32 k-chunks); thread owns 4 d-columns (float4 stream),
// partials accumulated with global_atomic_add_f32 into pre-zeroed agg.
// ---------------------------------------------------------------------------
#define KCH 32
#define KPER (KROWS / KCH)                  // 128
__global__ void wagg_kernel(const float* __restrict__ feat,
                            const float* __restrict__ weights,
                            float* __restrict__ agg) {
    __shared__ float wl[KPER];
    const int b = blockIdx.x, kch = blockIdx.y;
    const int d = threadIdx.x * 4;          // 256 threads * 4 = full D

    const float* wb = weights + (size_t)b * KROWS + kch * KPER;
    if (threadIdx.x < KPER) wl[threadIdx.x] = wb[threadIdx.x];
    __syncthreads();

    const float* fb = feat + ((size_t)b * KROWS + (size_t)kch * KPER) * D_DIM + d;
    float ax = 0.f, ay = 0.f, az = 0.f, aw = 0.f;
    #pragma unroll 4
    for (int k = 0; k < KPER; ++k) {
        const float4 f = *reinterpret_cast<const float4*>(fb + (size_t)k * D_DIM);
        const float w = wl[k];
        ax += f.x * w; ay += f.y * w; az += f.z * w; aw += f.w * w;
    }
    float* out = agg + b * D_DIM + d;
    atomicAdd(out + 0, ax);
    atomicAdd(out + 1, ay);
    atomicAdd(out + 2, az);
    atomicAdd(out + 3, aw);
}

// ---------------------------------------------------------------------------
extern "C" void kernel_launch(void* const* d_in, const int* in_sizes, int n_in,
                              void* d_out, int out_size, void* d_ws, size_t ws_size,
                              hipStream_t stream) {
    const float* feat = (const float*)d_in[0];
    const float* Wv   = (const float*)d_in[1];
    const float* bv   = (const float*)d_in[2];
    const float* Wu   = (const float*)d_in[3];
    const float* bu   = (const float*)d_in[4];
    const float* Ww   = (const float*)d_in[5];
    const float* bw   = (const float*)d_in[6];

    float* agg     = (float*)d_out;                    // [16,1024]
    float* weights = (float*)d_out + BATCH * D_DIM;    // [16,4096]

    unsigned short* wvS = (unsigned short*)d_ws;               // 512 KB
    unsigned short* wuS = wvS + W_ELEMS;                       // 512 KB
    float* logits = (float*)((char*)d_ws + 2 * W_ELEMS * 2);   // 256 KB

    // 0: weight swizzle (f32->bf16 frag layout) + zero aggregated region
    prep_kernel<<<(2 * W_ELEMS) / 256, 256, 0, stream>>>(Wv, Wu, wvS, wuS, agg);

    // 1: WMMA GEMM + gate + projection -> logits (128 KB dynamic LDS)
    gemm_logits_kernel<<<ROWS_TOTAL / ROWS_PER_BLOCK, 256, GEMM_SMEM_BYTES, stream>>>(
        feat, wvS, wuS, bv, bu, Ww, bw, logits);

    // 2: softmax over instances
    softmax_kernel<<<BATCH, 256, 0, stream>>>(logits, weights);

    // 3: weighted feature aggregation
    dim3 g(BATCH, KCH);
    wagg_kernel<<<g, 256, 0, stream>>>(feat, weights, agg);
}
